// Contrast_9122510537044
// MI455X (gfx1250) — matrix operand order repaired and checked by reading, not compiled, not run on hardware
//
#include <hip/hip_runtime.h>
#include <hip/hip_bf16.h>

// ---------------------------------------------------------------------------
// GRACE contrastive loss, fused: normalize -> 4 bf16 WMMA GEMMs with
// exp-rowsum epilogue (no N x N materialization) -> log/sum reduction.
// Target: gfx1250 (CDNA5), wave32, v_wmma_f32_16x16x32_bf16.
// ---------------------------------------------------------------------------

typedef __attribute__((ext_vector_type(16))) __bf16 v16bf;
typedef __attribute__((ext_vector_type(8)))  float  v8f;

#define N_ROWS 8192
#define DIM    512
// exp(x / 0.2) == exp2(x * log2(e) / 0.2)
#define EXP_SCALE 7.213475204444817f
#define INV_TAU   5.0f
#define EXP_ONE_OVER_TAU 148.4131591025766f   // exp(1/tau) = exp(5)

// ---------------------------------------------------------------------------
// Kernel 1: per-row L2 normalize z1,z2 -> bf16 h1,h2; diag[i] = h1_i . h2_i (fp32)
// ---------------------------------------------------------------------------
__global__ __launch_bounds__(256) void normalize_kernel(
    const float* __restrict__ z1, const float* __restrict__ z2,
    __hip_bfloat16* __restrict__ h1, __hip_bfloat16* __restrict__ h2,
    float* __restrict__ diag) {
  const int row = blockIdx.x;
  const int t   = threadIdx.x;
  const float* r1 = z1 + (size_t)row * DIM;
  const float* r2 = z2 + (size_t)row * DIM;
  const float a0 = r1[t], a1 = r1[t + 256];
  const float b0 = r2[t], b1 = r2[t + 256];

  __shared__ float s1[256], s2[256], s3[256];
  s1[t] = a0 * a0 + a1 * a1;
  s2[t] = b0 * b0 + b1 * b1;
  s3[t] = a0 * b0 + a1 * b1;
  __syncthreads();
  #pragma unroll
  for (int off = 128; off > 0; off >>= 1) {
    if (t < off) { s1[t] += s1[t + off]; s2[t] += s2[t + off]; s3[t] += s3[t + off]; }
    __syncthreads();
  }
  const float inv1 = rsqrtf(fmaxf(s1[0], 1e-24f));  // eps=1e-12 on the norm
  const float inv2 = rsqrtf(fmaxf(s2[0], 1e-24f));

  h1[(size_t)row * DIM + t]       = __float2bfloat16(a0 * inv1);
  h1[(size_t)row * DIM + t + 256] = __float2bfloat16(a1 * inv1);
  h2[(size_t)row * DIM + t]       = __float2bfloat16(b0 * inv2);
  h2[(size_t)row * DIM + t + 256] = __float2bfloat16(b1 * inv2);
  if (t == 0) diag[row] = s3[0] * inv1 * inv2;
}

// ---------------------------------------------------------------------------
// Kernel 0: zero the four row-sum accumulators
// ---------------------------------------------------------------------------
__global__ void zero_kernel(float* __restrict__ p, int n) {
  const int i = blockIdx.x * blockDim.x + threadIdx.x;
  if (i < n) p[i] = 0.0f;
}

// ---------------------------------------------------------------------------
// WMMA helpers: fragment loads matching the CDNA5 16-bit A / B VGPR layouts
// (CDNA5 ISA §7.12.2), sourced from row-major [N, DIM] bf16.
// A 16x32: lane l<16 holds row M=l, K = {0..7 (v0-3), 16..23 (v4-7)};
//          lane l>=16 holds row M=l-16, K = {8..15, 24..31}.
// B 32x16: lane l<16 holds col N=l, K = {0..7, 16..23};
//          lane l>=16 holds col N=l-16, K = {8..15, 24..31}.
// ---------------------------------------------------------------------------
union FragU { v16bf v; uint4 q[2]; };

__device__ __forceinline__ v16bf load_a_frag(const __hip_bfloat16* __restrict__ H,
                                             int row0, int k0, int lane) {
  const int m    = lane & 15;
  const int koff = (lane >> 4) << 3;   // 0 or 8
  const __hip_bfloat16* p = H + (size_t)(row0 + m) * DIM + k0 + koff;
  FragU f;
  f.q[0] = *reinterpret_cast<const uint4*>(p);        // K = koff .. koff+7
  f.q[1] = *reinterpret_cast<const uint4*>(p + 16);   // K = koff+16 .. koff+23
  return f.v;
}

__device__ __forceinline__ v16bf load_b_frag(const __hip_bfloat16* __restrict__ H,
                                             int col0, int k0, int lane) {
  const int n    = lane & 15;
  const int koff = (lane >> 4) << 3;   // 0 or 8
  const __hip_bfloat16* p = H + (size_t)(col0 + n) * DIM + k0 + koff;
  FragU f;
  f.q[0] = *reinterpret_cast<const uint4*>(p);        // K = koff .. koff+7
  f.q[1] = *reinterpret_cast<const uint4*>(p + 16);   // K = koff+16 .. koff+23
  return f.v;
}

__device__ __forceinline__ v8f wmma_bf16(v16bf a, v16bf b, v8f c) {
  // (neg_a, A, neg_b, B, c_mod, C, reuse_a, reuse_b)
  return __builtin_amdgcn_wmma_f32_16x16x32_bf16(false, a, false, b,
                                                 (short)0, c, false, false);
}

// ---------------------------------------------------------------------------
// Kernel 2: one of four products S = A x B^T (both row-major [N,DIM] bf16),
// fused epilogue out[i] += sum_j exp(S[i,j]/tau).
// Grid: (N/128, 1, 4). Block: 256 (8 waves). Wave tile: 32 rows x 64 cols.
// ---------------------------------------------------------------------------
__global__ __launch_bounds__(256) void simrow_kernel(
    const __hip_bfloat16* __restrict__ h1, const __hip_bfloat16* __restrict__ h2,
    float* __restrict__ rs11, float* __restrict__ rs22,
    float* __restrict__ rs12r, float* __restrict__ rs12c) {
  const __hip_bfloat16* A;
  const __hip_bfloat16* B;
  float* out;
  switch (blockIdx.z) {
    case 0:  A = h1; B = h1; out = rs11;  break;  // l1 refl rowsums
    case 1:  A = h2; B = h2; out = rs22;  break;  // l2 refl rowsums
    case 2:  A = h1; B = h2; out = rs12r; break;  // l1 between rowsums
    default: A = h2; B = h1; out = rs12c; break;  // l2 between rowsums
  }
  const int lane = threadIdx.x & 31;
  const int wave = threadIdx.x >> 5;   // 0..7
  const int wm   = wave & 3;           // 4 waves along M
  const int wn   = wave >> 2;          // 2 waves along N
  const int i0   = blockIdx.x * 128 + wm * 32;   // this wave's 32 rows

  v8f acc0 = {};   // per-lane partial rowsums, subtile rows i0..i0+15
  v8f acc1 = {};   // subtile rows i0+16..i0+31

  for (int j0 = 0; j0 < N_ROWS; j0 += 128) {
    const int jb = j0 + wn * 64;       // this wave's 64 columns
    v8f c[2][4];
    #pragma unroll
    for (int mi = 0; mi < 2; ++mi)
      #pragma unroll
      for (int ni = 0; ni < 4; ++ni) c[mi][ni] = v8f{};

    for (int k0 = 0; k0 < DIM; k0 += 32) {
      const v16bf a0 = load_a_frag(A, i0,      k0, lane);
      const v16bf a1 = load_a_frag(A, i0 + 16, k0, lane);
      const v16bf b0 = load_b_frag(B, jb,      k0, lane);
      const v16bf b1 = load_b_frag(B, jb + 16, k0, lane);
      const v16bf b2 = load_b_frag(B, jb + 32, k0, lane);
      const v16bf b3 = load_b_frag(B, jb + 48, k0, lane);
      c[0][0] = wmma_bf16(a0, b0, c[0][0]);
      c[0][1] = wmma_bf16(a0, b1, c[0][1]);
      c[0][2] = wmma_bf16(a0, b2, c[0][2]);
      c[0][3] = wmma_bf16(a0, b3, c[0][3]);
      c[1][0] = wmma_bf16(a1, b0, c[1][0]);
      c[1][1] = wmma_bf16(a1, b1, c[1][1]);
      c[1][2] = wmma_bf16(a1, b2, c[1][2]);
      c[1][3] = wmma_bf16(a1, b3, c[1][3]);
    }
    // epilogue: exp(S/tau) elementwise, fold into per-lane row partials
    #pragma unroll
    for (int ni = 0; ni < 4; ++ni) {
      #pragma unroll
      for (int r = 0; r < 8; ++r) {
        acc0[r] += __builtin_amdgcn_exp2f(c[0][ni][r] * EXP_SCALE);
        acc1[r] += __builtin_amdgcn_exp2f(c[1][ni][r] * EXP_SCALE);
      }
    }
  }

  // C layout: VGPR r, lanes 0-15 -> row r (cols = lane); lanes 16-31 -> row r+8.
  // Reduce the 16 column-lanes of each half (xor masks stay within halves).
  #pragma unroll
  for (int r = 0; r < 8; ++r) {
    float v0 = acc0[r], v1 = acc1[r];
    #pragma unroll
    for (int m = 8; m >= 1; m >>= 1) {
      v0 += __shfl_xor(v0, m, 32);
      v1 += __shfl_xor(v1, m, 32);
    }
    acc0[r] = v0;
    acc1[r] = v1;
  }
  if (lane == 0 || lane == 16) {
    const int mb = (lane >> 4) * 8;    // lane 0 -> rows r, lane 16 -> rows r+8
    #pragma unroll
    for (int r = 0; r < 8; ++r) {
      atomicAdd(&out[i0 + mb + r],      acc0[r]);   // two wn-waves combine here
      atomicAdd(&out[i0 + 16 + mb + r], acc1[r]);
    }
  }
}

// ---------------------------------------------------------------------------
// Kernel 3: loss = sum_i [ -diag_i/tau + 0.5*(log d1_i + log d2_i) ]
// ---------------------------------------------------------------------------
__global__ __launch_bounds__(256) void loss_kernel(
    const float* __restrict__ diag,
    const float* __restrict__ rs11, const float* __restrict__ rs22,
    const float* __restrict__ rs12r, const float* __restrict__ rs12c,
    float* __restrict__ out) {
  __shared__ float sh[256];
  float s = 0.0f;
  for (int i = threadIdx.x; i < N_ROWS; i += 256) {
    const float d1 = rs11[i] + rs12r[i] - EXP_ONE_OVER_TAU;
    const float d2 = rs22[i] + rs12c[i] - EXP_ONE_OVER_TAU;
    s += -diag[i] * INV_TAU + 0.5f * (__logf(d1) + __logf(d2));
  }
  sh[threadIdx.x] = s;
  __syncthreads();
  #pragma unroll
  for (int off = 128; off > 0; off >>= 1) {
    if (threadIdx.x < off) sh[threadIdx.x] += sh[threadIdx.x + off];
    __syncthreads();
  }
  if (threadIdx.x == 0) out[0] = sh[0];
}

// ---------------------------------------------------------------------------
extern "C" void kernel_launch(void* const* d_in, const int* in_sizes, int n_in,
                              void* d_out, int out_size, void* d_ws, size_t ws_size,
                              hipStream_t stream) {
  const float* z1 = (const float*)d_in[0];
  const float* z2 = (const float*)d_in[1];

  char* ws = (char*)d_ws;
  const size_t hbytes = (size_t)N_ROWS * DIM * sizeof(__hip_bfloat16);  // 8 MB
  __hip_bfloat16* h1 = (__hip_bfloat16*)ws;
  __hip_bfloat16* h2 = (__hip_bfloat16*)(ws + hbytes);
  float* diag  = (float*)(ws + 2 * hbytes);
  float* rs11  = diag  + N_ROWS;
  float* rs22  = rs11  + N_ROWS;
  float* rs12r = rs22  + N_ROWS;
  float* rs12c = rs12r + N_ROWS;

  zero_kernel<<<(4 * N_ROWS + 255) / 256, 256, 0, stream>>>(rs11, 4 * N_ROWS);
  normalize_kernel<<<N_ROWS, 256, 0, stream>>>(z1, z2, h1, h2, diag);

  dim3 grid(N_ROWS / 128, 1, 4);
  simrow_kernel<<<grid, 256, 0, stream>>>(h1, h2, rs11, rs22, rs12r, rs12c);

  loss_kernel<<<1, 256, 0, stream>>>(diag, rs11, rs22, rs12r, rs12c, (float*)d_out);
}
